// OrbitGNN_44324062495090
// MI455X (gfx1250) — compile-verified, matrix operand order checked by reading, not run on hardware
//
#include <hip/hip_runtime.h>
#include <hip/hip_fp16.h>

// ---------------------------------------------------------------------------
// OrbitGNN (2-layer GAT + MLP head) for MI455X / gfx1250, wave32 + WMMA.
// GEMMs: v_wmma_f32_16x16x32_f16 with LDS-staged B fragments (ds_load path).
// Edge softmax + aggregation: L2-resident gathers + global float atomics.
// ---------------------------------------------------------------------------

typedef __attribute__((ext_vector_type(16))) _Float16 v16h;
typedef __attribute__((ext_vector_type(8)))  _Float16 v8h;
typedef __attribute__((ext_vector_type(8)))  float    v8f;

#define GN_N 100000
#define GN_E 1600000

// ---------------------------------------------------------------------------
// Utility fills (graph-capture safe; no hipMemset)
// ---------------------------------------------------------------------------
__global__ void fill_u32_kernel(unsigned int* __restrict__ p, unsigned int v, long long n) {
    long long i = (long long)blockIdx.x * blockDim.x + threadIdx.x;
    if (i < n) p[i] = v;
}

// x [N,7] f32 -> x16 [N,32] f16 zero-padded (K padded to one WMMA K-step)
__global__ void pack_x_kernel(const float* __restrict__ x, _Float16* __restrict__ x16, int N) {
    int i = blockIdx.x * blockDim.x + threadIdx.x;
    if (i >= N * 32) return;
    int n = i >> 5, c = i & 31;
    float v = (c < 7) ? x[n * 7 + c] : 0.f;
    x16[i] = (_Float16)v;
}

// Pre-swizzle B (f32 row-major [K,Ncols]) into per-lane WMMA fragment order:
// Bp[((nt*KSTEPS+ks)*32 + lane)*16 + i] = B[k][nt*16 + (lane&15)],
// k = ks*32 + (i<8 ? (lane>>4)*8+i : 16+(lane>>4)*8+(i-8)); zero-pad OOB.
__global__ void pack_b_kernel(const float* __restrict__ W, _Float16* __restrict__ Bp,
                              int K, int Ncols, int KSTEPS, int NT) {
    int t = blockIdx.x * blockDim.x + threadIdx.x;
    int total = NT * KSTEPS * 512;
    if (t >= total) return;
    int i    = t & 15;
    int lane = (t >> 4) & 31;
    int ks   = (t >> 9) % KSTEPS;
    int nt   = t / (KSTEPS * 512);
    int half_ = lane >> 4;
    int n = nt * 16 + (lane & 15);
    int k = ks * 32 + ((i < 8) ? (half_ * 8 + i) : (16 + half_ * 8 + (i - 8)));
    float v = (k < K && n < Ncols) ? W[k * Ncols + n] : 0.f;
    Bp[t] = (_Float16)v;
}

// ---------------------------------------------------------------------------
// WMMA GEMM: D = act(A[M, KSTEPS*32] (f16) x Bp + bias)
// One wave per 16-row M-tile; 4 waves / block share B staged in LDS.
// All output config is compile-time -> branchless immediate-offset stores,
// and the bias add is elided entirely when HAS_BIAS=false (x+0.0f is not
// foldable under strict FP semantics).
//   DF_COLS > 0 : f32 output with row stride DF_COLS
//   DH_COLS > 0 : f16 output with row stride DH_COLS
//   NSTORE      : number of valid output columns (< NT*16 only for final layer)
// ---------------------------------------------------------------------------
template <int KSTEPS, int NT, bool RELU, bool HAS_BIAS, int DF_COLS, int DH_COLS, int NSTORE>
__global__ void wmma_gemm_kernel(const _Float16* __restrict__ A,
                                 const _Float16* __restrict__ Bp,
                                 const float* __restrict__ bias,
                                 float* __restrict__ Df,
                                 _Float16* __restrict__ Dh,
                                 int M) {
    const int K = KSTEPS * 32;
    __shared__ _Float16 Bs[NT * KSTEPS * 512];

    // Cooperative B stage (before any wave exits; barrier is unconditional).
    {
        const uint4* gsrc = (const uint4*)Bp;
        uint4* ldst = (uint4*)Bs;
        const int nvec = NT * KSTEPS * 64;   // total halfs / 8
#pragma unroll 2
        for (int i = threadIdx.x; i < nvec; i += 128) ldst[i] = gsrc[i];
    }
    __syncthreads();

    int gtid = blockIdx.x * blockDim.x + threadIdx.x;
    int wave = gtid >> 5;
    int lane = gtid & 31;
    int m0 = wave * 16;
    if (m0 >= M) return;                 // wave-uniform: EXEC stays all-1s
    int half_ = lane >> 4;
    int mrow  = m0 + (lane & 15);

    // A fragments: per lane, per k-step, two contiguous 16B global loads.
    v16h afrag[KSTEPS];
#pragma unroll
    for (int ks = 0; ks < KSTEPS; ++ks) {
        const _Float16* ap = A + (size_t)mrow * K + ks * 32 + half_ * 8;
        v8h lo = *(const v8h*)ap;          // K = half*8 .. +7
        v8h hi = *(const v8h*)(ap + 16);   // K = 16+half*8 .. +7
        v16h a;
#pragma unroll
        for (int j = 0; j < 8; ++j) { a[j] = lo[j]; a[8 + j] = hi[j]; }
        afrag[ks] = a;
    }

    int ncolBase = lane & 15;
#pragma unroll
    for (int nt = 0; nt < NT; ++nt) {
        v8f acc = {};
#pragma unroll
        for (int ks = 0; ks < KSTEPS; ++ks) {
            v16h b = *((const v16h*)Bs + ((nt * KSTEPS + ks) * 32 + lane));  // ds_load
            acc = __builtin_amdgcn_wmma_f32_16x16x32_f16(
                false, afrag[ks], false, b, (short)0, acc, false, false);
        }
        int ncol = nt * 16 + ncolBase;
        bool valid = (NSTORE >= NT * 16) || (ncol < NSTORE);
        if (valid) {
            float bv = HAS_BIAS ? bias[ncol] : 0.f;
            int rowBase = m0 + half_ * 8;    // D layout: VGPR r -> row half*8+r
            if (DF_COLS > 0) {
                float* p = Df + (size_t)rowBase * DF_COLS + ncol;
#pragma unroll
                for (int r = 0; r < 8; ++r) {
                    float v = HAS_BIAS ? (acc[r] + bv) : acc[r];
                    if (RELU) v = fmaxf(v, 0.f);
                    p[r * DF_COLS] = v;      // immediate-offset stores
                }
            }
            if (DH_COLS > 0) {
                _Float16* p = Dh + (size_t)rowBase * DH_COLS + ncol;
#pragma unroll
                for (int r = 0; r < 8; ++r) {
                    float v = HAS_BIAS ? (acc[r] + bv) : acc[r];
                    if (RELU) v = fmaxf(v, 0.f);
                    p[r * DH_COLS] = (_Float16)v;
                }
            }
        }
    }
}

// ---------------------------------------------------------------------------
// Attention coefficients per node: asrc[n,h] = sum_c h[n,h,c]*a_src[h,c]
// ---------------------------------------------------------------------------
__global__ void alpha_kernel(const float* __restrict__ h,
                             const float* __restrict__ a_src,
                             const float* __restrict__ a_dst,
                             float* __restrict__ asrc, float* __restrict__ adst, int N) {
    int n = blockIdx.x * blockDim.x + threadIdx.x;
    if (n >= N) return;
    const float4* hp = (const float4*)(h + (size_t)n * 128);
    const float4* as = (const float4*)a_src;
    const float4* ad = (const float4*)a_dst;
    float s0 = 0.f, s1 = 0.f, d0 = 0.f, d1 = 0.f;
#pragma unroll
    for (int j = 0; j < 16; ++j) {
        float4 h0 = hp[j], h1 = hp[16 + j];
        float4 u = as[j], v = ad[j], u1 = as[16 + j], v1 = ad[16 + j];
        s0 += h0.x * u.x + h0.y * u.y + h0.z * u.z + h0.w * u.w;
        d0 += h0.x * v.x + h0.y * v.y + h0.z * v.z + h0.w * v.w;
        s1 += h1.x * u1.x + h1.y * u1.y + h1.z * u1.z + h1.w * u1.w;
        d1 += h1.x * v1.x + h1.y * v1.y + h1.z * v1.z + h1.w * v1.w;
    }
    asrc[2 * n] = s0; asrc[2 * n + 1] = s1;
    adst[2 * n] = d0; adst[2 * n + 1] = d1;
}

// IEEE-monotonic float atomic max (memory pre-initialized to -inf bits)
__device__ __forceinline__ void atomicMaxF(float* addr, float v) {
    if (v >= 0.f) atomicMax((int*)addr, __float_as_int(v));
    else          atomicMin((unsigned int*)addr, (unsigned int)__float_as_int(v));
}

// e = leaky_relu(asrc[src]+adst[tgt]); segment max by target
__global__ void edge_score_kernel(const int* __restrict__ src, const int* __restrict__ tgt,
                                  const float* __restrict__ asrc, const float* __restrict__ adst,
                                  float* __restrict__ e, float* __restrict__ mbuf, int E) {
    int idx = blockIdx.x * blockDim.x + threadIdx.x;
    if (idx >= E) return;
    int s = src[idx], t = tgt[idx];
    float2 av = *(const float2*)(asrc + 2 * s);
    float2 dv = *(const float2*)(adst + 2 * t);
    float v0 = av.x + dv.x, v1 = av.y + dv.y;
    v0 = (v0 > 0.f) ? v0 : 0.2f * v0;
    v1 = (v1 > 0.f) ? v1 : 0.2f * v1;
    *(float2*)(e + 2 * idx) = make_float2(v0, v1);
    atomicMaxF(&mbuf[2 * t + 0], v0);
    atomicMaxF(&mbuf[2 * t + 1], v1);
}

// e <- exp(e - m[tgt]); segment sum by target
__global__ void edge_exp_kernel(const int* __restrict__ tgt, float* __restrict__ e,
                                const float* __restrict__ mbuf, float* __restrict__ sbuf, int E) {
    int idx = blockIdx.x * blockDim.x + threadIdx.x;
    if (idx >= E) return;
    int t = tgt[idx];
    float2 ev = *(const float2*)(e + 2 * idx);
    float2 mv = *(const float2*)(mbuf + 2 * t);
    float e0 = __expf(ev.x - mv.x), e1 = __expf(ev.y - mv.y);
    *(float2*)(e + 2 * idx) = make_float2(e0, e1);
    atomicAdd(&sbuf[2 * t + 0], e0);
    atomicAdd(&sbuf[2 * t + 1], e1);
}

// agg[tgt,:] += h[src,:] * alpha ; one thread = (edge, 4 channels)
__global__ void edge_agg_kernel(const int* __restrict__ src, const int* __restrict__ tgt,
                                const float* __restrict__ h, const float* __restrict__ e,
                                const float* __restrict__ sbuf, float* __restrict__ agg, int E) {
    long long gid = (long long)blockIdx.x * blockDim.x + threadIdx.x;
    if (gid >= (long long)E * 32) return;
    int idx = (int)(gid >> 5);
    int c0  = (int)(gid & 31) * 4;
    int s = src[idx], t = tgt[idx];
    int hd = c0 >> 6;
    const float* hp = h + (size_t)s * 128 + c0;
    __builtin_prefetch(hp, 0, 0);  // launch h[src] fetch while score/sum loads resolve
    float alpha = e[2 * idx + hd] / (sbuf[2 * t + hd] + 1e-16f);
    float4 hv = *(const float4*)hp;
    float* ap = agg + (size_t)t * 128 + c0;
    atomicAdd(ap + 0, hv.x * alpha);
    atomicAdd(ap + 1, hv.y * alpha);
    atomicAdd(ap + 2, hv.z * alpha);
    atomicAdd(ap + 3, hv.w * alpha);
}

// layer1: relu(agg + b1) -> f16 (concat=True, 128 ch)
__global__ void finalize1_kernel(const float* __restrict__ agg, const float* __restrict__ b1,
                                 _Float16* __restrict__ out16, int N) {
    int i = blockIdx.x * blockDim.x + threadIdx.x;
    if (i >= N * 128) return;
    int c = i & 127;
    float v = fmaxf(agg[i] + b1[c], 0.f);
    out16[i] = (_Float16)v;
}

// layer2: mean over heads + b2 -> f16 (concat=False, 64 ch, no relu)
__global__ void finalize2_kernel(const float* __restrict__ agg, const float* __restrict__ b2,
                                 _Float16* __restrict__ out16, int N) {
    int i = blockIdx.x * blockDim.x + threadIdx.x;
    if (i >= N * 64) return;
    int n = i >> 6, c = i & 63;
    float v = 0.5f * (agg[(size_t)n * 128 + c] + agg[(size_t)n * 128 + 64 + c]) + b2[c];
    out16[i] = (_Float16)v;
}

// ---------------------------------------------------------------------------
// Launch
// ---------------------------------------------------------------------------
extern "C" void kernel_launch(void* const* d_in, const int* in_sizes, int n_in,
                              void* d_out, int out_size, void* d_ws, size_t ws_size,
                              hipStream_t stream) {
    (void)in_sizes; (void)n_in; (void)out_size; (void)ws_size;

    const float* x      = (const float*)d_in[0];
    const int*   ei     = (const int*)d_in[1];
    const float* W1     = (const float*)d_in[2];
    const float* a_src1 = (const float*)d_in[3];
    const float* a_dst1 = (const float*)d_in[4];
    const float* b1     = (const float*)d_in[5];
    const float* W2     = (const float*)d_in[6];
    const float* a_src2 = (const float*)d_in[7];
    const float* a_dst2 = (const float*)d_in[8];
    const float* b2     = (const float*)d_in[9];
    const float* Wc1    = (const float*)d_in[10];
    const float* bc1    = (const float*)d_in[11];
    const float* Wc2    = (const float*)d_in[12];
    const float* bc2    = (const float*)d_in[13];
    float* out = (float*)d_out;

    const int N = GN_N, E = GN_E;
    const int* src = ei;
    const int* tgt = ei + E;

    // Workspace layout (256B aligned)
    char* ws = (char*)d_ws;
    size_t off = 0;
    auto alloc = [&](size_t bytes) -> char* {
        char* p = ws + off;
        off = (off + bytes + 255) & ~(size_t)255;
        return p;
    };
    _Float16* x16     = (_Float16*)alloc((size_t)N * 32 * 2);
    float*    h1      = (float*)   alloc((size_t)N * 128 * 4);  // reused as h2
    _Float16* out1_16 = (_Float16*)alloc((size_t)N * 128 * 2);
    _Float16* out2_16 = (_Float16*)alloc((size_t)N * 64 * 2);
    _Float16* z16     = (_Float16*)alloc((size_t)N * 64 * 2);
    float*    asrc    = (float*)   alloc((size_t)N * 2 * 4);
    float*    adst    = (float*)   alloc((size_t)N * 2 * 4);
    float*    mbuf    = (float*)   alloc((size_t)N * 2 * 4);
    float*    sbuf    = (float*)   alloc((size_t)N * 2 * 4);
    float*    ebuf    = (float*)   alloc((size_t)E * 2 * 4);
    float*    agg     = (float*)   alloc((size_t)N * 128 * 4);
    _Float16* W1p     = (_Float16*)alloc(8 * 1 * 512 * 2);
    _Float16* W2p     = (_Float16*)alloc(8 * 4 * 512 * 2);
    _Float16* Wc1p    = (_Float16*)alloc(4 * 2 * 512 * 2);
    _Float16* Wc2p    = (_Float16*)alloc(1 * 2 * 512 * 2);

    const dim3 B256(256);
    const int gemmBlocks = ((N / 16) + 3) / 4;   // 4 waves / 128-thread block

    // Pack weights + input features to f16 fragment layouts
    pack_b_kernel<<<(8 * 1 * 512 + 255) / 256, B256, 0, stream>>>(W1,  W1p,  7,   128, 1, 8);
    pack_b_kernel<<<(8 * 4 * 512 + 255) / 256, B256, 0, stream>>>(W2,  W2p,  128, 128, 4, 8);
    pack_b_kernel<<<(4 * 2 * 512 + 255) / 256, B256, 0, stream>>>(Wc1, Wc1p, 64,  64,  2, 4);
    pack_b_kernel<<<(1 * 2 * 512 + 255) / 256, B256, 0, stream>>>(Wc2, Wc2p, 64,  4,   2, 1);
    pack_x_kernel<<<(N * 32 + 255) / 256, B256, 0, stream>>>(x, x16, N);

    // ---- GAT layer 1 ----
    wmma_gemm_kernel<1, 8, false, false, 128, 0, 128><<<gemmBlocks, 128, 0, stream>>>(
        x16, W1p, nullptr, h1, nullptr, N);
    alpha_kernel<<<(N + 255) / 256, B256, 0, stream>>>(h1, a_src1, a_dst1, asrc, adst, N);
    fill_u32_kernel<<<(2 * N + 255) / 256, B256, 0, stream>>>((unsigned*)mbuf, 0xFF800000u, 2 * N);
    fill_u32_kernel<<<(2 * N + 255) / 256, B256, 0, stream>>>((unsigned*)sbuf, 0u, 2 * N);
    fill_u32_kernel<<<(N * 128 + 255) / 256, B256, 0, stream>>>((unsigned*)agg, 0u, (long long)N * 128);
    edge_score_kernel<<<(E + 255) / 256, B256, 0, stream>>>(src, tgt, asrc, adst, ebuf, mbuf, E);
    edge_exp_kernel<<<(E + 255) / 256, B256, 0, stream>>>(tgt, ebuf, mbuf, sbuf, E);
    edge_agg_kernel<<<(int)(((long long)E * 32 + 255) / 256), B256, 0, stream>>>(
        src, tgt, h1, ebuf, sbuf, agg, E);
    finalize1_kernel<<<(N * 128 + 255) / 256, B256, 0, stream>>>(agg, b1, out1_16, N);

    // ---- GAT layer 2 (reuse h buffer) ----
    float* h2 = h1;
    wmma_gemm_kernel<4, 8, false, false, 128, 0, 128><<<gemmBlocks, 128, 0, stream>>>(
        out1_16, W2p, nullptr, h2, nullptr, N);
    alpha_kernel<<<(N + 255) / 256, B256, 0, stream>>>(h2, a_src2, a_dst2, asrc, adst, N);
    fill_u32_kernel<<<(2 * N + 255) / 256, B256, 0, stream>>>((unsigned*)mbuf, 0xFF800000u, 2 * N);
    fill_u32_kernel<<<(2 * N + 255) / 256, B256, 0, stream>>>((unsigned*)sbuf, 0u, 2 * N);
    fill_u32_kernel<<<(N * 128 + 255) / 256, B256, 0, stream>>>((unsigned*)agg, 0u, (long long)N * 128);
    edge_score_kernel<<<(E + 255) / 256, B256, 0, stream>>>(src, tgt, asrc, adst, ebuf, mbuf, E);
    edge_exp_kernel<<<(E + 255) / 256, B256, 0, stream>>>(tgt, ebuf, mbuf, sbuf, E);
    edge_agg_kernel<<<(int)(((long long)E * 32 + 255) / 256), B256, 0, stream>>>(
        src, tgt, h2, ebuf, sbuf, agg, E);
    finalize2_kernel<<<(N * 64 + 255) / 256, B256, 0, stream>>>(agg, b2, out2_16, N);

    // ---- MLP head ----
    wmma_gemm_kernel<2, 4, true, true, 0, 64, 64><<<gemmBlocks, 128, 0, stream>>>(
        out2_16, Wc1p, bc1, nullptr, z16, N);
    wmma_gemm_kernel<2, 1, false, true, 4, 0, 4><<<gemmBlocks, 128, 0, stream>>>(
        z16, Wc2p, bc2, out, nullptr, N);
}
